// FFNwMoE_11278584119789
// MI455X (gfx1250) — matrix operand order repaired and checked by reading, not compiled, not run on hardware
//
#include <hip/hip_runtime.h>
#include <hip/hip_bf16.h>
#include <math.h>

typedef __bf16 bf16_t;
typedef __attribute__((ext_vector_type(16))) __bf16 v16bf;
typedef __attribute__((ext_vector_type(8)))  __bf16 v8bf;
typedef __attribute__((ext_vector_type(8)))  float  v8f;

constexpr int T = 4096, D = 1024, H = 2048, E = 8;
constexpr float AUX_COEF = 0.01f;

#define WMMA_BF16(A, B, C) \
  __builtin_amdgcn_wmma_f32_16x16x32_bf16(false, (A), false, (B), (short)0, (C), false, false)

__device__ __forceinline__ float gelu_exact(float v) {
  return 0.5f * v * (1.0f + erff(v * 0.70710678118654752f));
}

// A fragment (16x32 bf16, contiguous k order used here for A and B): lane holds row lane&15,
// K {0..15} (lanes 0-15) or {16..31} (lanes 16-31).
__device__ __forceinline__ v16bf load_afrag(const bf16_t* __restrict__ row, int kb, int ka) {
  v8bf lo = *(const v8bf*)(row + kb + 2 * ka);
  v8bf hi = *(const v8bf*)(row + kb + 2 * ka + 8);
  v16bf a;
#pragma unroll
  for (int i = 0; i < 8; ++i) { a[i] = lo[i]; a[8 + i] = hi[i]; }
  return a;
}

// ---------------- init: zero output + per-expert counters ----------------
__global__ void init_k(float* out, int n_out, int* count) {
  int i = blockIdx.x * blockDim.x + threadIdx.x;
  for (int j = i; j < n_out; j += gridDim.x * blockDim.x) out[j] = 0.f;
  if (i < E) count[i] = 0;
}

// ---------------- fp32 -> bf16 conversion ----------------
__global__ void cvt_k(const float* __restrict__ src, bf16_t* __restrict__ dst, int n) {
  int stride = gridDim.x * blockDim.x;
  for (int i = blockIdx.x * blockDim.x + threadIdx.x; i < n; i += stride)
    dst[i] = (bf16_t)src[i];
}

// ---------------- router: logits, softmax, top-2, list build ----------------
__global__ void router_k(const float* __restrict__ x, const float* __restrict__ Wr,
                         int* list, float* gw, float* probs, int* top1, int* count) {
  int t = blockIdx.x * blockDim.x + threadIdx.x;
  if (t >= T) return;
  const float* xr = x + (size_t)t * D;
  float acc[E];
#pragma unroll
  for (int e = 0; e < E; ++e) acc[e] = 0.f;
  for (int d = 0; d < D; d += 4) {
    float4 xv = *(const float4*)(xr + d);
#pragma unroll
    for (int e = 0; e < E; ++e) {
      const float* wr = Wr + (size_t)e * D + d;
      acc[e] += xv.x * wr[0] + xv.y * wr[1] + xv.z * wr[2] + xv.w * wr[3];
    }
  }
  float m = acc[0];
#pragma unroll
  for (int e = 1; e < E; ++e) m = fmaxf(m, acc[e]);
  float p[E]; float s = 0.f;
#pragma unroll
  for (int e = 0; e < E; ++e) { p[e] = expf(acc[e] - m); s += p[e]; }
  float inv = 1.f / s;
#pragma unroll
  for (int e = 0; e < E; ++e) { p[e] *= inv; probs[(size_t)t * E + e] = p[e]; }
  int i1 = 0;
#pragma unroll
  for (int e = 1; e < E; ++e) if (acc[e] > acc[i1]) i1 = e;
  int i2 = (i1 == 0) ? 1 : 0;
#pragma unroll
  for (int e = 0; e < E; ++e) if (e != i1 && acc[e] > acc[i2]) i2 = e;
  top1[t] = i1;
  int pos1 = atomicAdd(&count[i1], 1);
  list[i1 * T + pos1] = t; gw[i1 * T + pos1] = p[i1];
  int pos2 = atomicAdd(&count[i2], 1);
  list[i2 * T + pos2] = t; gw[i2 * T + pos2] = p[i2];
}

// ---------------- finalize: aux loss + prefix offsets ----------------
__global__ void finalize_k(const float* __restrict__ probs, const int* __restrict__ top1,
                           const int* __restrict__ count, int* offset, float* aux_out) {
  __shared__ float sP[256][E];
  __shared__ int   sC[256][E];
  int tid = threadIdx.x;
  float lp[E]; int lc[E];
#pragma unroll
  for (int e = 0; e < E; ++e) { lp[e] = 0.f; lc[e] = 0; }
  for (int t = tid; t < T; t += 256) {
#pragma unroll
    for (int e = 0; e < E; ++e) lp[e] += probs[(size_t)t * E + e];
    lc[top1[t]]++;
  }
#pragma unroll
  for (int e = 0; e < E; ++e) { sP[tid][e] = lp[e]; sC[tid][e] = lc[e]; }
  __syncthreads();
  for (int off = 128; off > 0; off >>= 1) {
    if (tid < off) {
#pragma unroll
      for (int e = 0; e < E; ++e) { sP[tid][e] += sP[tid + off][e]; sC[tid][e] += sC[tid + off][e]; }
    }
    __syncthreads();
  }
  if (tid == 0) {
    float aux = 0.f;
#pragma unroll
    for (int e = 0; e < E; ++e)
      aux += ((float)sC[0][e] / (float)T) * (sP[0][e] / (float)T);
    aux_out[0] = AUX_COEF * (float)E * aux;
    int off = 0;
#pragma unroll
    for (int e = 0; e < E; ++e) { offset[e] = off; off += count[e]; }
    offset[E] = off;
  }
}

// ---------------- pass 1: h = gelu(x@w1^T) * (x@w3^T), bf16 WMMA ----------------
// Block = 8 waves covering 64 rows x 128 h-cols. Each wave: 32 rows x 32 cols,
// 8 WMMA per K-step: 2 A fragments reused across 2 col tiles x {w1, w3};
// each B fragment reused across 2 row tiles (halves weight-stream L2 traffic).
__global__ void __launch_bounds__(256)
up_gate_k(const bf16_t* __restrict__ xb, const bf16_t* __restrict__ w1b,
          const bf16_t* __restrict__ w3b, bf16_t* __restrict__ hbuf,
          const int* __restrict__ list, const int* __restrict__ count,
          const int* __restrict__ offset) {
  const int e = blockIdx.z;
  const int cnt = (e == E) ? T : count[e];
  const int mBase = blockIdx.x * 64;
  if (mBase >= cnt) return;
  const int wave = threadIdx.x >> 5;
  const int lane = threadIdx.x & 31;
  const int half = lane >> 4;
  const int nl   = lane & 15;
  const int rowSlot0 = mBase + (wave >> 2) * 32;       // 32 rows per wave
  const int n0 = blockIdx.y * 128 + (wave & 3) * 32;   // first 16-col tile
  const int n1 = n0 + 16;                              // second 16-col tile
  if (rowSlot0 >= cnt) return;                         // wave-uniform exit
  int sA0 = rowSlot0 + nl;       if (sA0 >= cnt) sA0 = cnt - 1;
  int sA1 = rowSlot0 + 16 + nl;  if (sA1 >= cnt) sA1 = cnt - 1;
  const int tok0 = (e == E) ? sA0 : list[e * T + sA0];
  const int tok1 = (e == E) ? sA1 : list[e * T + sA1];
  const bf16_t* xrow0  = xb + (size_t)tok0 * D;
  const bf16_t* xrow1  = xb + (size_t)tok1 * D;
  const bf16_t* w1rowA = w1b + ((size_t)e * H + (n0 + nl)) * D + half * 16;
  const bf16_t* w1rowB = w1b + ((size_t)e * H + (n1 + nl)) * D + half * 16;
  const bf16_t* w3rowA = w3b + ((size_t)e * H + (n0 + nl)) * D + half * 16;
  const bf16_t* w3rowB = w3b + ((size_t)e * H + (n1 + nl)) * D + half * 16;
  const int ka = half << 3;
  v8f c1a0 = {}, c1b0 = {}, c3a0 = {}, c3b0 = {};
  v8f c1a1 = {}, c1b1 = {}, c3a1 = {}, c3b1 = {};
  for (int kb = 0; kb < D; kb += 32) {
    v16bf a0  = load_afrag(xrow0, kb, ka);
    v16bf a1  = load_afrag(xrow1, kb, ka);
    v16bf b1a = *(const v16bf*)(w1rowA + kb);
    v16bf b1b = *(const v16bf*)(w1rowB + kb);
    v16bf b3a = *(const v16bf*)(w3rowA + kb);
    v16bf b3b = *(const v16bf*)(w3rowB + kb);
    __builtin_prefetch(w1rowA + kb + 128, 0, 3);
    __builtin_prefetch(w3rowA + kb + 128, 0, 3);
    c1a0 = WMMA_BF16(a0, b1a, c1a0);
    c1a1 = WMMA_BF16(a1, b1a, c1a1);
    c1b0 = WMMA_BF16(a0, b1b, c1b0);
    c1b1 = WMMA_BF16(a1, b1b, c1b1);
    c3a0 = WMMA_BF16(a0, b3a, c3a0);
    c3a1 = WMMA_BF16(a1, b3a, c3a1);
    c3b0 = WMMA_BF16(a0, b3b, c3b0);
    c3b1 = WMMA_BF16(a1, b3b, c3b1);
  }
  const size_t rowGBase = (size_t)((e == E) ? 2 * T : offset[e]);
#pragma unroll
  for (int v = 0; v < 8; ++v) {
    int slot0 = rowSlot0 + v + half * 8;
    if (slot0 < cnt) {
      bf16_t* hrow = hbuf + (rowGBase + slot0) * (size_t)H;
      hrow[n0 + nl] = (bf16_t)(gelu_exact(c1a0[v]) * c3a0[v]);
      hrow[n1 + nl] = (bf16_t)(gelu_exact(c1b0[v]) * c3b0[v]);
    }
    int slot1 = slot0 + 16;
    if (slot1 < cnt) {
      bf16_t* hrow = hbuf + (rowGBase + slot1) * (size_t)H;
      hrow[n0 + nl] = (bf16_t)(gelu_exact(c1a1[v]) * c3a1[v]);
      hrow[n1 + nl] = (bf16_t)(gelu_exact(c1b1[v]) * c3b1[v]);
    }
  }
}

// ---------------- pass 2: out += gate * (h @ w2^T), bf16 WMMA ----------------
// Block = 8 waves covering 64 rows x 128 d-cols. Each wave: 32 rows x 32 cols,
// 4 WMMA per K-step; B fragments reused across 2 row tiles.
__global__ void __launch_bounds__(256)
down_k(const bf16_t* __restrict__ hbuf, const bf16_t* __restrict__ w2b,
       float* __restrict__ out, const int* __restrict__ list,
       const float* __restrict__ gw, const int* __restrict__ count,
       const int* __restrict__ offset) {
  const int e = blockIdx.z;
  const int cnt = (e == E) ? T : count[e];
  const int mBase = blockIdx.x * 64;
  if (mBase >= cnt) return;
  const int wave = threadIdx.x >> 5;
  const int lane = threadIdx.x & 31;
  const int half = lane >> 4;
  const int nl   = lane & 15;
  const int rowSlot0 = mBase + (wave >> 2) * 32;
  const int n0 = blockIdx.y * 128 + (wave & 3) * 32;
  const int n1 = n0 + 16;
  if (rowSlot0 >= cnt) return;
  const size_t rowGBase = (size_t)((e == E) ? 2 * T : offset[e]);
  int sA0 = rowSlot0 + nl;       if (sA0 >= cnt) sA0 = cnt - 1;
  int sA1 = rowSlot0 + 16 + nl;  if (sA1 >= cnt) sA1 = cnt - 1;
  const bf16_t* hrow0  = hbuf + (rowGBase + sA0) * (size_t)H;
  const bf16_t* hrow1  = hbuf + (rowGBase + sA1) * (size_t)H;
  const bf16_t* w2rowA = w2b + ((size_t)e * D + (n0 + nl)) * (size_t)H + half * 16;
  const bf16_t* w2rowB = w2b + ((size_t)e * D + (n1 + nl)) * (size_t)H + half * 16;
  const int ka = half << 3;
  v8f cA0 = {}, cB0 = {}, cA1 = {}, cB1 = {};
  for (int kb = 0; kb < H; kb += 32) {
    v16bf a0 = load_afrag(hrow0, kb, ka);
    v16bf a1 = load_afrag(hrow1, kb, ka);
    v16bf bA = *(const v16bf*)(w2rowA + kb);
    v16bf bB = *(const v16bf*)(w2rowB + kb);
    __builtin_prefetch(w2rowA + kb + 128, 0, 3);
    cA0 = WMMA_BF16(a0, bA, cA0);
    cA1 = WMMA_BF16(a1, bA, cA1);
    cB0 = WMMA_BF16(a0, bB, cB0);
    cB1 = WMMA_BF16(a1, bB, cB1);
  }
#pragma unroll
  for (int v = 0; v < 8; ++v) {
    int slot0 = rowSlot0 + v + half * 8;
    if (slot0 < cnt) {
      int   token = (e == E) ? slot0 : list[e * T + slot0];
      float g     = (e == E) ? 1.0f : gw[e * T + slot0];
      float* orow = out + (size_t)token * D;
      atomicAdd(orow + n0 + nl, g * cA0[v]);
      atomicAdd(orow + n1 + nl, g * cB0[v]);
    }
    int slot1 = slot0 + 16;
    if (slot1 < cnt) {
      int   token = (e == E) ? slot1 : list[e * T + slot1];
      float g     = (e == E) ? 1.0f : gw[e * T + slot1];
      float* orow = out + (size_t)token * D;
      atomicAdd(orow + n0 + nl, g * cA1[v]);
      atomicAdd(orow + n1 + nl, g * cB1[v]);
    }
  }
}

extern "C" void kernel_launch(void* const* d_in, const int* in_sizes, int n_in,
                              void* d_out, int out_size, void* d_ws, size_t ws_size,
                              hipStream_t stream) {
  (void)in_sizes; (void)n_in; (void)out_size; (void)ws_size;
  const float* x   = (const float*)d_in[0];
  const float* Wr  = (const float*)d_in[1];
  const float* w1  = (const float*)d_in[2];
  const float* w2  = (const float*)d_in[3];
  const float* w3  = (const float*)d_in[4];
  const float* w1s = (const float*)d_in[5];
  const float* w2s = (const float*)d_in[6];
  const float* w3s = (const float*)d_in[7];
  float* out = (float*)d_out;

  // workspace layout (~165 MiB): bf16 weights (9 experts incl. shared) +
  // bf16 x + packed bf16 h buffer (3T rows) + routing lists + aux scratch.
  char* ws = (char*)d_ws;
  const size_t HD  = (size_t)H * D;             // 2,097,152 elems
  const size_t W9B = 9 * HD * 2;                // bytes per bf16 weight block
  bf16_t* w1b  = (bf16_t*)(ws);
  bf16_t* w3b  = (bf16_t*)(ws + W9B);
  bf16_t* w2b  = (bf16_t*)(ws + 2 * W9B);
  bf16_t* xbuf = (bf16_t*)(ws + 3 * W9B);       // T x D bf16
  bf16_t* hbuf = (bf16_t*)(ws + 3 * W9B + (size_t)T * D * 2);  // 3T x H bf16
  char* p = ws + 3 * W9B + (size_t)T * D * 2 + (size_t)3 * T * H * 2;
  int*   list   = (int*)p;   p += (size_t)E * T * 4;
  float* gwv    = (float*)p; p += (size_t)E * T * 4;
  float* probs  = (float*)p; p += (size_t)T * E * 4;
  int*   top1   = (int*)p;   p += (size_t)T * 4;
  int*   count  = (int*)p;   p += 64;
  int*   offset = (int*)p;   p += 64;
  (void)p;

  init_k<<<1024, 256, 0, stream>>>(out, T * D + 1, count);

  cvt_k<<<2048, 256, 0, stream>>>(w1,  w1b,          (int)(E * HD));
  cvt_k<<<256,  256, 0, stream>>>(w1s, w1b + 8 * HD, (int)HD);
  cvt_k<<<2048, 256, 0, stream>>>(w3,  w3b,          (int)(E * HD));
  cvt_k<<<256,  256, 0, stream>>>(w3s, w3b + 8 * HD, (int)HD);
  cvt_k<<<2048, 256, 0, stream>>>(w2,  w2b,          (int)(E * HD));
  cvt_k<<<256,  256, 0, stream>>>(w2s, w2b + 8 * HD, (int)HD);
  cvt_k<<<1024, 256, 0, stream>>>(x,   xbuf,         T * D);

  router_k<<<T / 256, 256, 0, stream>>>(x, Wr, list, gwv, probs, top1, count);
  finalize_k<<<1, 256, 0, stream>>>(probs, top1, count, offset, out + (size_t)T * D);

  dim3 g1(T / 64, H / 128, E + 1);
  up_gate_k<<<g1, 256, 0, stream>>>(xbuf, w1b, w3b, hbuf, list, count, offset);

  dim3 g2(T / 64, D / 128, E + 1);
  down_k<<<g2, 256, 0, stream>>>(hbuf, w2b, out, list, gwv, count, offset);
}
